// MultiHeadAttention_27496380629133
// MI455X (gfx1250) — compile-verified, matrix operand order checked by reading, not run on hardware
//
#include <hip/hip_runtime.h>
#include <math.h>

typedef __bf16 bf16;
typedef __attribute__((ext_vector_type(16))) __bf16 v16bf;
typedef __attribute__((ext_vector_type(8)))  float  v8f;

union V16 {
  v16bf v;
  uint4 q[2];
};

#define BB   2
#define LL   2048
#define DD   256
#define HH   8
#define DKV  32

static __device__ __forceinline__ v8f wmma_bf16(const V16& a, const V16& b, v8f c) {
  return __builtin_amdgcn_wmma_f32_16x16x32_bf16(false, a.v, false, b.v,
                                                 (short)0, c, false, false);
}

// ---------------------------------------------------------------------------
// Kernel 1: transpose + bf16-pack the four weight matrices: wt[n][k] = w[k][n]
// Fold 1/sqrt(dk) into w_qs.
// ---------------------------------------------------------------------------
__global__ void mha_pack_w(const float* __restrict__ wq, const float* __restrict__ wk,
                           const float* __restrict__ wv, const float* __restrict__ wfc,
                           bf16* __restrict__ wtq, bf16* __restrict__ wtk,
                           bf16* __restrict__ wtv, bf16* __restrict__ wtfc) {
  const int idx = blockIdx.x * 256 + threadIdx.x;   // 4 * 65536 threads total
  const int mat = idx >> 16;
  const int r   = idx & 65535;
  const int n   = r >> 8, k = r & 255;
  const float* src = (mat == 0) ? wq : (mat == 1) ? wk : (mat == 2) ? wv : wfc;
  bf16*        dst = (mat == 0) ? wtq : (mat == 1) ? wtk : (mat == 2) ? wtv : wtfc;
  float s = src[k * 256 + n];
  if (mat == 0) s *= 0.17677669529663689f;  // 1/sqrt(32)
  dst[n * 256 + k] = (bf16)s;
}

// ---------------------------------------------------------------------------
// Kernel 2: projections. [B*L,256] x [256,256] bf16 WMMA GEMM.
// z=0 -> qh [b,h,l,32], z=1 -> kh [b,h,l,32], z=2 -> vt [b,h,32,l] (transposed)
// ---------------------------------------------------------------------------
__global__ void __launch_bounds__(256)
mha_proj(const float* __restrict__ q, const float* __restrict__ k,
         const float* __restrict__ v,
         const bf16* __restrict__ wtq, const bf16* __restrict__ wtk,
         const bf16* __restrict__ wtv,
         bf16* __restrict__ qh, bf16* __restrict__ kh, bf16* __restrict__ vt) {
  const int z = blockIdx.y;
  const float* x  = (z == 0) ? q : (z == 1) ? k : v;
  const bf16*  wt = (z == 0) ? wtq : (z == 1) ? wtk : wtv;

  __shared__ alignas(16) bf16 X[16 * 256];
  const int tid = threadIdx.x;
  const int mtile = blockIdx.x;
  #pragma unroll
  for (int i = 0; i < 16; i++)
    X[i * 256 + tid] = (bf16)x[(size_t)(mtile * 16 + i) * 256 + tid];
  __syncthreads();

  const int wave = tid >> 5, lane = tid & 31, half = lane >> 4, lm = lane & 15;
  const int koff = half * 8;
  for (int s = 0; s < 2; s++) {
    const int nbase = (wave * 2 + s) * 16;
    v8f acc = {};
    #pragma unroll
    for (int kk = 0; kk < 8; kk++) {
      const int kb = kk * 32;
      V16 a, bm;
      a.q[0] = *(const uint4*)&X[lm * 256 + kb + koff];
      a.q[1] = *(const uint4*)&X[lm * 256 + kb + koff + 16];
      const bf16* wp = wt + (nbase + lm) * 256 + kb;
      bm.q[0] = *(const uint4*)(wp + koff);
      bm.q[1] = *(const uint4*)(wp + koff + 16);
      acc = wmma_bf16(a, bm, acc);
    }
    #pragma unroll
    for (int j = 0; j < 8; j++) {
      const int row = half * 8 + j;
      const int gr  = mtile * 16 + row;          // global row in [0, B*L)
      const int b   = gr >> 11, l = gr & 2047;
      const int n   = nbase + lm;
      const int h   = n >> 5, d = n & 31;
      if (z < 2) {
        bf16* dst = (z == 0) ? qh : kh;
        dst[(((size_t)(b * HH + h) * LL + l) << 5) + d] = (bf16)acc[j];
      } else {
        vt[(((size_t)(b * HH + h) * DKV + d) << 11) + l] = (bf16)acc[j];
      }
    }
  }
}

// ---------------------------------------------------------------------------
// Kernel 3: attention. One WG = (b, h, 32-query tile); each of the 8 waves owns
// a 256-key stripe. Pass 1: QK^T WMMA + online (max,sum) softmax stats.
// Pass 2: recompute QK^T, write normalized probs to global attn output, and
// immediately accumulate P@V with bf16 WMMA (P round-tripped through LDS to
// transpose C-layout -> A-layout).
// ---------------------------------------------------------------------------
__global__ void __launch_bounds__(256)
mha_attn(const bf16* __restrict__ qh, const bf16* __restrict__ kh,
         const bf16* __restrict__ vt,
         float* __restrict__ attn, bf16* __restrict__ oh) {
  const int tid = threadIdx.x;
  const int wave = tid >> 5, lane = tid & 31, half = lane >> 4, lm = lane & 15;
  const int koff = half * 8;
  const int bh = blockIdx.z * HH + blockIdx.y;
  const int qb = blockIdx.x * 32;

  __shared__ float Sm[8][32];
  __shared__ float Sl[8][32];
  __shared__ float gM[32];
  __shared__ float gI[32];
  __shared__ alignas(16) bf16 P[8][32][32];
  __shared__ float Ored[8][32][32];

  // Q A-tiles (rows 0..15 and 16..31 of the query tile), dk=32 = one WMMA K
  V16 aq0, aq1;
  {
    const bf16* p0 = qh + (size_t)(bh * LL + qb + lm) * DKV;
    aq0.q[0] = *(const uint4*)(p0 + koff);
    aq0.q[1] = *(const uint4*)(p0 + koff + 16);
    const bf16* p1 = qh + (size_t)(bh * LL + qb + 16 + lm) * DKV;
    aq1.q[0] = *(const uint4*)(p1 + koff);
    aq1.q[1] = *(const uint4*)(p1 + koff + 16);
  }

  const int kstart = wave * 256;

  // ---- pass 1: online row max / expsum over this wave's 256 keys ----
  float pm[16], pl[16];
  #pragma unroll
  for (int j = 0; j < 16; j++) { pm[j] = -1e30f; pl[j] = 0.f; }

  for (int t = 0; t < 16; t++) {
    const int kb = kstart + t * 16;
    const bf16* kp = kh + (size_t)(bh * LL + kb + lm) * DKV;
    __builtin_prefetch(kp + 16 * DKV, 0, 1);  // next 16-key tile
    V16 bk;
    bk.q[0] = *(const uint4*)(kp + koff);
    bk.q[1] = *(const uint4*)(kp + koff + 16);
    v8f z = {};
    v8f c0 = wmma_bf16(aq0, bk, z);
    v8f c1 = wmma_bf16(aq1, bk, z);
    #pragma unroll
    for (int j = 0; j < 8; j++) {
      float s0 = c0[j];
      if (s0 > pm[j]) { pl[j] = pl[j] * __expf(pm[j] - s0) + 1.f; pm[j] = s0; }
      else            { pl[j] += __expf(s0 - pm[j]); }
      float s1 = c1[j];
      if (s1 > pm[8 + j]) { pl[8 + j] = pl[8 + j] * __expf(pm[8 + j] - s1) + 1.f; pm[8 + j] = s1; }
      else                { pl[8 + j] += __expf(s1 - pm[8 + j]); }
    }
  }
  // butterfly reduce across the 16 lanes holding the same rows
  #pragma unroll
  for (int mask = 1; mask < 16; mask <<= 1) {
    #pragma unroll
    for (int j = 0; j < 16; j++) {
      float om = __shfl_xor(pm[j], mask, 32);
      float ol = __shfl_xor(pl[j], mask, 32);
      float nm = fmaxf(pm[j], om);
      pl[j] = pl[j] * __expf(pm[j] - nm) + ol * __expf(om - nm);
      pm[j] = nm;
    }
  }
  if (lm == 0) {
    #pragma unroll
    for (int j = 0; j < 8; j++) {
      Sm[wave][half * 8 + j]      = pm[j];     Sl[wave][half * 8 + j]      = pl[j];
      Sm[wave][16 + half * 8 + j] = pm[8 + j]; Sl[wave][16 + half * 8 + j] = pl[8 + j];
    }
  }
  __syncthreads();
  if (tid < 32) {
    float m = -1e30f;
    #pragma unroll
    for (int w = 0; w < 8; w++) m = fmaxf(m, Sm[w][tid]);
    float l = 0.f;
    #pragma unroll
    for (int w = 0; w < 8; w++) l += Sl[w][tid] * __expf(Sm[w][tid] - m);
    gM[tid] = m;
    gI[tid] = 1.0f / l;
  }
  __syncthreads();

  float M0[8], I0[8], M1[8], I1[8];
  #pragma unroll
  for (int j = 0; j < 8; j++) {
    M0[j] = gM[half * 8 + j];      I0[j] = gI[half * 8 + j];
    M1[j] = gM[16 + half * 8 + j]; I1[j] = gI[16 + half * 8 + j];
  }

  // ---- pass 2: recompute scores, emit probs, accumulate P@V ----
  v8f o00 = {}, o01 = {}, o10 = {}, o11 = {};
  for (int kc = 0; kc < 8; kc++) {
    const int kb = kstart + kc * 32;
    #pragma unroll
    for (int sub = 0; sub < 2; sub++) {
      const int kbs = kb + sub * 16;
      const bf16* kp = kh + (size_t)(bh * LL + kbs + lm) * DKV;
      V16 bk;
      bk.q[0] = *(const uint4*)(kp + koff);
      bk.q[1] = *(const uint4*)(kp + koff + 16);
      v8f z = {};
      v8f c0 = wmma_bf16(aq0, bk, z);
      v8f c1 = wmma_bf16(aq1, bk, z);
      #pragma unroll
      for (int j = 0; j < 8; j++) {
        const int r0 = half * 8 + j;
        float p0 = __expf(c0[j] - M0[j]) * I0[j];
        attn[((size_t)(bh * LL + qb + r0)) * LL + kbs + lm] = p0;
        P[wave][r0][sub * 16 + lm] = (bf16)p0;
        const int r1 = 16 + half * 8 + j;
        float p1 = __expf(c1[j] - M1[j]) * I1[j];
        attn[((size_t)(bh * LL + qb + r1)) * LL + kbs + lm] = p1;
        P[wave][r1][sub * 16 + lm] = (bf16)p1;
      }
    }
    // Rebuild P as A-tiles (LDS ops within one wave are in-order; P region is
    // private to this wave).
    V16 ap0, ap1, bv0, bv1;
    ap0.q[0] = *(const uint4*)&P[wave][lm][koff];
    ap0.q[1] = *(const uint4*)&P[wave][lm][koff + 16];
    ap1.q[0] = *(const uint4*)&P[wave][16 + lm][koff];
    ap1.q[1] = *(const uint4*)&P[wave][16 + lm][koff + 16];
    // V^T rows are contiguous in key -> clean 16B B-tile loads
    const bf16* vp0 = vt + ((size_t)(bh * DKV + lm)) * LL + kb;
    bv0.q[0] = *(const uint4*)(vp0 + koff);
    bv0.q[1] = *(const uint4*)(vp0 + koff + 16);
    const bf16* vp1 = vt + ((size_t)(bh * DKV + 16 + lm)) * LL + kb;
    bv1.q[0] = *(const uint4*)(vp1 + koff);
    bv1.q[1] = *(const uint4*)(vp1 + koff + 16);
    o00 = wmma_bf16(ap0, bv0, o00);
    o01 = wmma_bf16(ap0, bv1, o01);
    o10 = wmma_bf16(ap1, bv0, o10);
    o11 = wmma_bf16(ap1, bv1, o11);
  }

  // reduce O partials across the 8 waves
  #pragma unroll
  for (int j = 0; j < 8; j++) {
    Ored[wave][half * 8 + j][lm]           = o00[j];
    Ored[wave][half * 8 + j][16 + lm]      = o01[j];
    Ored[wave][16 + half * 8 + j][lm]      = o10[j];
    Ored[wave][16 + half * 8 + j][16 + lm] = o11[j];
  }
  __syncthreads();
  const int b = blockIdx.z, h = blockIdx.y;
  #pragma unroll
  for (int e0 = 0; e0 < 4; e0++) {
    int e = tid * 4 + e0;
    int m = e >> 5, n = e & 31;
    float s = 0.f;
    #pragma unroll
    for (int w = 0; w < 8; w++) s += Ored[w][m][n];
    oh[((size_t)(b * LL + qb + m)) * DD + h * DKV + n] = (bf16)s;
  }
}

// ---------------------------------------------------------------------------
// Kernel 4: out = LN( oh @ w_fc + residual )
// ---------------------------------------------------------------------------
__global__ void __launch_bounds__(256)
mha_out(const bf16* __restrict__ oh, const bf16* __restrict__ wtfc,
        const float* __restrict__ resid,
        const float* __restrict__ gamma, const float* __restrict__ beta,
        float* __restrict__ out) {
  __shared__ alignas(16) bf16 A[16 * 256];
  __shared__ float Obuf[16][256];
  __shared__ float redS[16][16];
  __shared__ float redQ[16][16];
  __shared__ float muA[16], rvA[16];
  const int tid = threadIdx.x, mtile = blockIdx.x;
  #pragma unroll
  for (int i = 0; i < 16; i++)
    A[i * 256 + tid] = oh[(size_t)(mtile * 16 + i) * 256 + tid];
  __syncthreads();

  const int wave = tid >> 5, lane = tid & 31, half = lane >> 4, lm = lane & 15;
  const int koff = half * 8;
  for (int s = 0; s < 2; s++) {
    const int nbase = (wave * 2 + s) * 16;
    v8f acc = {};
    #pragma unroll
    for (int kk = 0; kk < 8; kk++) {
      const int kb = kk * 32;
      V16 a, bm;
      a.q[0] = *(const uint4*)&A[lm * 256 + kb + koff];
      a.q[1] = *(const uint4*)&A[lm * 256 + kb + koff + 16];
      const bf16* wp = wtfc + (nbase + lm) * 256 + kb;
      bm.q[0] = *(const uint4*)(wp + koff);
      bm.q[1] = *(const uint4*)(wp + koff + 16);
      acc = wmma_bf16(a, bm, acc);
    }
    #pragma unroll
    for (int j = 0; j < 8; j++) {
      const int row = half * 8 + j;
      const int gr  = mtile * 16 + row;
      const int n   = nbase + lm;
      Obuf[row][n] = acc[j] + resid[(size_t)gr * 256 + n];
    }
  }
  __syncthreads();
  {
    const int r = tid >> 4, j0 = tid & 15;
    float s1 = 0.f, s2 = 0.f;
    #pragma unroll
    for (int i = 0; i < 16; i++) {
      float x = Obuf[r][j0 + 16 * i];
      s1 += x; s2 += x * x;
    }
    redS[r][j0] = s1; redQ[r][j0] = s2;
  }
  __syncthreads();
  if (tid < 16) {
    float s1 = 0.f, s2 = 0.f;
    #pragma unroll
    for (int i = 0; i < 16; i++) { s1 += redS[tid][i]; s2 += redQ[tid][i]; }
    float mu  = s1 * (1.0f / 256.0f);
    float var = s2 * (1.0f / 256.0f) - mu * mu;
    muA[tid] = mu;
    rvA[tid] = rsqrtf(var + 1e-6f);
  }
  __syncthreads();
  const float g = gamma[tid], be = beta[tid];
  #pragma unroll
  for (int r = 0; r < 16; r++) {
    const int gr = mtile * 16 + r;
    out[(size_t)gr * 256 + tid] = (Obuf[r][tid] - muA[r]) * rvA[r] * g + be;
  }
}

// ---------------------------------------------------------------------------
extern "C" void kernel_launch(void* const* d_in, const int* in_sizes, int n_in,
                              void* d_out, int out_size, void* d_ws, size_t ws_size,
                              hipStream_t stream) {
  (void)in_sizes; (void)n_in; (void)out_size; (void)ws_size;
  const float* q   = (const float*)d_in[0];
  const float* k   = (const float*)d_in[1];
  const float* v   = (const float*)d_in[2];
  const float* wq  = (const float*)d_in[3];
  const float* wk  = (const float*)d_in[4];
  const float* wv  = (const float*)d_in[5];
  const float* wfc = (const float*)d_in[6];
  const float* g   = (const float*)d_in[7];
  const float* be  = (const float*)d_in[8];

  float* out  = (float*)d_out;
  float* attn = out + (size_t)BB * LL * DD;   // tuple output: out, then attn

  bf16* ws   = (bf16*)d_ws;
  bf16* wtq  = ws;
  bf16* wtk  = wtq + 65536;
  bf16* wtv  = wtk + 65536;
  bf16* wtfc = wtv + 65536;
  bf16* qh   = wtfc + 65536;
  bf16* kh   = qh + (size_t)BB * HH * LL * DKV;
  bf16* vt   = kh + (size_t)BB * HH * LL * DKV;
  bf16* oh   = vt + (size_t)BB * HH * LL * DKV;

  mha_pack_w<<<1024, 256, 0, stream>>>(wq, wk, wv, wfc, wtq, wtk, wtv, wtfc);
  mha_proj<<<dim3(256, 3), 256, 0, stream>>>(q, k, v, wtq, wtk, wtv, qh, kh, vt);
  mha_attn<<<dim3(LL / 32, HH, BB), 256, 0, stream>>>(qh, kh, vt, attn, oh);
  mha_out<<<256, 256, 0, stream>>>(oh, wtfc, q, g, be, out);
}
